// OriginFloatLinear_56375740727556
// MI455X (gfx1250) — 8-bit matrix operand order repaired and checked by reading, compiled, not run on hardware
//
#include <hip/hip_runtime.h>
#include <hip/hip_bf16.h>


typedef __attribute__((ext_vector_type(16))) int   v16i;
typedef __attribute__((ext_vector_type(8)))  float v8f;
typedef int v4i_gcc __attribute__((vector_size(16)));

#define GLOBAL_AS __attribute__((address_space(1)))
#define LDS_AS    __attribute__((address_space(3)))

#if defined(__AMDGCN__) && __has_builtin(__builtin_amdgcn_global_load_async_to_lds_b128)
#define ASYNC_LDS 1
#else
#define ASYNC_LDS 0
#endif

__device__ __forceinline__ void copy64B_to_lds(const unsigned char* g, unsigned char* l) {
#if ASYNC_LDS
    GLOBAL_AS v4i_gcc* gp = (GLOBAL_AS v4i_gcc*)g;
    LDS_AS    v4i_gcc* lp = (LDS_AS    v4i_gcc*)l;
    __builtin_amdgcn_global_load_async_to_lds_b128(gp, lp, 0,  0);
    __builtin_amdgcn_global_load_async_to_lds_b128(gp, lp, 16, 0);
    __builtin_amdgcn_global_load_async_to_lds_b128(gp, lp, 32, 0);
    __builtin_amdgcn_global_load_async_to_lds_b128(gp, lp, 48, 0);
#else
    const uint4* g4 = (const uint4*)g;
    uint4*       l4 = (uint4*)l;
    l4[0] = g4[0]; l4[1] = g4[1]; l4[2] = g4[2]; l4[3] = g4[3];
#endif
}

__device__ __forceinline__ void wait_async_lds() {
#if ASYNC_LDS
#if __has_builtin(__builtin_amdgcn_s_wait_asynccnt)
    __builtin_amdgcn_s_wait_asynccnt(0);
#else
    asm volatile("s_wait_asynccnt 0x0" ::: "memory");
#endif
#endif
}

__device__ __forceinline__ unsigned int f32_to_e4m3(float xf) {
    unsigned int u    = __float_as_uint(xf);
    unsigned int sign = (u >> 24) & 0x80u;
    unsigned int au   = u & 0x7FFFFFFFu;
    if (au >= 0x7F800000u) return sign | 0x7Fu;
    float a = __uint_as_float(au);
    if (a >= 448.0f) return sign | 0x7Eu;
    if (a < 0.015625f) {
        int m = (int)rintf(a * 512.0f);
        return sign | (unsigned int)m;
    }
    int e = (int)((au >> 23) & 0xFFu) - 127;
    float rs = __uint_as_float((unsigned int)((130 - e) << 23));
    int m = (int)rintf(a * rs);
    if (m == 16) { ++e; m = 8; }
    return sign | (unsigned int)(((e + 7) << 3) | (m - 8));
}

__global__ __launch_bounds__(256)
void quant_x_kernel(const float* __restrict__ x, unsigned char* __restrict__ xq,
                    float* __restrict__ sx, long long nBlocks) {
    long long blk = (long long)blockIdx.x * 8 + (threadIdx.x >> 5);
    if (blk >= nBlocks) return;
    int lane = threadIdx.x & 31;
    const float4 f = *(const float4*)(x + blk * 128 + lane * 4);
    float amax = fmaxf(fmaxf(fabsf(f.x), fabsf(f.y)), fmaxf(fabsf(f.z), fabsf(f.w)));
    #pragma unroll
    for (int off = 16; off > 0; off >>= 1)
        amax = fmaxf(amax, __shfl_xor(amax, off, 32));
    float scale = fmaxf(amax, 1e-12f) * (1.0f / 448.0f);
    unsigned int packed =  f32_to_e4m3(f.x / scale)
                        | (f32_to_e4m3(f.y / scale) << 8)
                        | (f32_to_e4m3(f.z / scale) << 16)
                        | (f32_to_e4m3(f.w / scale) << 24);
    *(unsigned int*)(xq + blk * 128 + lane * 4) = packed;
    if (lane == 0) sx[blk] = scale;
}

__global__ __launch_bounds__(256)
void quant_w_kernel(const float* __restrict__ w, unsigned char* __restrict__ wq,
                    long long n4) {
    long long i = (long long)blockIdx.x * 256 + threadIdx.x;
    if (i >= n4) return;
    const float4 f = ((const float4*)w)[i];
    unsigned int packed =  f32_to_e4m3(f.x)
                        | (f32_to_e4m3(f.y) << 8)
                        | (f32_to_e4m3(f.z) << 16)
                        | (f32_to_e4m3(f.w) << 24);
    ((unsigned int*)wq)[i] = packed;
}

__global__ __launch_bounds__(256)
void fp8_gemm_kernel(const unsigned char* __restrict__ xq,
                     const unsigned char* __restrict__ wq,
                     const float* __restrict__ sx,
                     const float* __restrict__ wsinv,
                     const float* __restrict__ bias,
                     __hip_bfloat16* __restrict__ out,
                     int M, int N, int K) {
    const int Kb    = K >> 7;
    const int nb    = blockIdx.x;
    const int mb    = blockIdx.y;
    const int mbase = mb << 7;
    const int nbase = nb << 7;

    __shared__ __align__(16) unsigned char As[2][128 * 128];
    __shared__ __align__(16) unsigned char Bs[2][128 * 128];
    __shared__ float sxs[2][128];

    const int tid  = threadIdx.x;
    const int lane = tid & 31;
    const int wave = tid >> 5;
    const int wm   = wave >> 1;
    const int wn   = wave & 1;
    const int hi   = (lane >> 4) & 1;
    const int l16  = lane & 15;

    const int sr = tid >> 1;
    const int sc = (tid & 1) << 6;
    const unsigned char* gA = xq + (size_t)(mbase + sr) * K + sc;
    const unsigned char* gB = wq + (size_t)(nbase + sr) * K + sc;
    unsigned char* lA[2] = { &As[0][sr * 128 + sc], &As[1][sr * 128 + sc] };
    unsigned char* lB[2] = { &Bs[0][sr * 128 + sc], &Bs[1][sr * 128 + sc] };

    v8f acc[2][4];
    #pragma unroll
    for (int i = 0; i < 2; ++i)
        #pragma unroll
        for (int j = 0; j < 4; ++j)
            #pragma unroll
            for (int v = 0; v < 8; ++v) acc[i][j][v] = 0.0f;

    copy64B_to_lds(gA, lA[0]);
    copy64B_to_lds(gB, lB[0]);
    if (tid < 128) sxs[0][tid] = sx[(size_t)(mbase + tid) * Kb];
    wait_async_lds();
    __syncthreads();

    for (int kb = 0; kb < Kb; ++kb) {
        const int cur = kb & 1;
        const int nxt = cur ^ 1;

        if (kb + 1 < Kb) {
            const size_t koff = (size_t)(kb + 1) << 7;
            copy64B_to_lds(gA + koff, lA[nxt]);
            copy64B_to_lds(gB + koff, lB[nxt]);
            if (tid < 128) sxs[nxt][tid] = sx[(size_t)(mbase + tid) * Kb + kb + 1];
        }

        const float wsb = wsinv[(size_t)nb * Kb + kb];

        v16i afrag[2];
        float cs[2][8];
        #pragma unroll
        for (int i = 0; i < 2; ++i) {
            const int row = wm * 32 + i * 16 + l16;
            const unsigned char* ap = &As[cur][row * 128 + hi * 8];
            union { v16i v; unsigned long long q[8]; } ua;
            #pragma unroll
            for (int t = 0; t < 8; ++t)
                ua.q[t] = *(const unsigned long long*)(ap + t * 16);
            afrag[i] = ua.v;
            #pragma unroll
            for (int v = 0; v < 8; ++v)
                cs[i][v] = sxs[cur][wm * 32 + i * 16 + hi * 8 + v] * wsb;
        }

        #pragma unroll
        for (int j = 0; j < 4; ++j) {
            const int col = wn * 64 + j * 16 + l16;
            const unsigned char* bp = &Bs[cur][col * 128 + hi * 8];
            union { v16i v; unsigned long long q[8]; } ub;
            #pragma unroll
            for (int g = 0; g < 8; ++g)
                ub.q[g] = *(const unsigned long long*)(bp + g * 16);
            #pragma unroll
            for (int i = 0; i < 2; ++i) {
                v8f zero = {0.f,0.f,0.f,0.f,0.f,0.f,0.f,0.f};
                v8f d = __builtin_amdgcn_wmma_f32_16x16x128_fp8_fp8(
                            afrag[i], ub.v, (short)0, zero, false, false);
                #pragma unroll
                for (int v = 0; v < 8; ++v)
                    acc[i][j][v] += d[v] * cs[i][v];
            }
        }

        if (kb + 1 < Kb) wait_async_lds();
        __syncthreads();
    }

    #pragma unroll
    for (int j = 0; j < 4; ++j) {
        const int n = nbase + wn * 64 + j * 16 + l16;
        const float bv = bias[n];
        #pragma unroll
        for (int i = 0; i < 2; ++i) {
            #pragma unroll
            for (int v = 0; v < 8; ++v) {
                const int m = mbase + wm * 32 + i * 16 + hi * 8 + v;
                out[(size_t)m * N + n] = __float2bfloat16(acc[i][j][v] + bv);
            }
        }
    }
}

extern "C" void kernel_launch(void* const* d_in, const int* in_sizes, int n_in,
                              void* d_out, int out_size, void* d_ws, size_t ws_size,
                              hipStream_t stream) {
    const float* x     = (const float*)d_in[0];
    const float* w     = (const float*)d_in[1];
    const float* wsinv = (const float*)d_in[2];
    const float* bias  = (const float*)d_in[3];
    __hip_bfloat16* out = (__hip_bfloat16*)d_out;

    const long long xk = in_sizes[0];
    const long long wk = in_sizes[1];
    const int N  = in_sizes[3];
    const int K  = (int)(wk / N);
    const int M  = (int)(xk / K);
    const int Kb = K / 128;

    unsigned char* xq = (unsigned char*)d_ws;
    unsigned char* wq = xq + xk;
    float* sx         = (float*)(wq + wk);

    const long long nBlocksX = (long long)M * Kb;
    quant_x_kernel<<<(unsigned)((nBlocksX + 7) / 8), 256, 0, stream>>>(x, xq, sx, nBlocksX);

    const long long n4 = wk / 4;
    quant_w_kernel<<<(unsigned)((n4 + 255) / 256), 256, 0, stream>>>(w, wq, n4);

    dim3 grid((unsigned)(N / 128), (unsigned)(M / 128));
    fp8_gemm_kernel<<<grid, 256, 0, stream>>>(xq, wq, sx, wsinv, bias, out, M, N, K);
}
